// SelfAttention_65867618451577
// MI455X (gfx1250) — matrix operand order repaired and checked by reading, not compiled, not run on hardware
//
#include <hip/hip_runtime.h>
#include <hip/hip_bf16.h>

// Problem constants (from reference): B=32, C=128, L=2048, Cd=C/2=64
#define B_  32
#define C_  128
#define CD  64
#define L_  2048

typedef __attribute__((ext_vector_type(16))) _Float16 v16h;
typedef __attribute__((ext_vector_type(8)))  float    v8f;
typedef int v4i_t __attribute__((vector_size(16)));   // builtin's pointee type

// ---- gfx1250 async global->LDS path (guarded; fallback = staged copy) ----
#if defined(__has_builtin)
# if __has_builtin(__builtin_amdgcn_global_load_async_to_lds_b128)
#  define HAVE_ASYNC_LDS 1
# endif
#endif
#ifndef HAVE_ASYNC_LDS
# define HAVE_ASYNC_LDS 0
#endif

#if HAVE_ASYNC_LDS
# if __has_builtin(__builtin_amdgcn_s_wait_asynccnt)
#  define ASYNC_WAIT(N) __builtin_amdgcn_s_wait_asynccnt(N)
# else
#  define ASYNC_WAIT(N) asm volatile("s_wait_asynccnt %0" :: "n"(N) : "memory")
# endif
#else
# define ASYNC_WAIT(N) ((void)0)
#endif

// Copy one 16-byte chunk global -> LDS (async when available).
__device__ __forceinline__ void cp16_g2l(const _Float16* __restrict__ g,
                                         _Float16* __restrict__ l) {
#if HAVE_ASYNC_LDS
    __builtin_amdgcn_global_load_async_to_lds_b128((v4i_t*)g, (v4i_t*)l, 0, 0);
#else
    *(uint4*)l = *(const uint4*)g;
#endif
}

// -------------------------------------------------------------------------
// Kernel 1: fused projections  K=Wq*x, Q=Wk*x, V=Wv*x, XV=Wv1*x
// Stacked output rows [0,64)=K, [64,128)=Q, [128,192)=V, [192,256)=XV.
// Q,K stored (B, L, 64) f16 ; V stored (B, 64, L) f16 ; XV (B, 64, L) f32.
// Grid: (L/128, B), block 256 (8 waves). 512 v_wmma_f32_16x16x32_f16 / WG.
// -------------------------------------------------------------------------
__global__ __launch_bounds__(256) void proj_kernel(
    const float* __restrict__ x,
    const float* __restrict__ Wq, const float* __restrict__ Wk,
    const float* __restrict__ Wv, const float* __restrict__ Wv1,
    _Float16* __restrict__ Kh, _Float16* __restrict__ Qh,
    _Float16* __restrict__ Vh, float* __restrict__ XV)
{
    constexpr int LDX = 136;                  // padded row (f16) to spread banks
    __shared__ __align__(16) _Float16 Xs[128 * LDX]; // x^T tile [l][c], 34 KB

    const int b   = blockIdx.y;
    const int l0  = blockIdx.x * 128;
    const int tid = threadIdx.x;

    // Stage x tile (C=128 x 128 cols) into LDS transposed + converted to f16.
    for (int i = 0; i < 64; ++i) {
        int idx = tid + i * 256;              // 0..16383
        int c   = idx >> 7;
        int ll  = idx & 127;
        float v = x[((size_t)b * C_ + c) * L_ + (l0 + ll)];
        Xs[ll * LDX + c] = (_Float16)v;
    }
    __syncthreads();

    const int lane   = tid & 31;
    const int lane16 = lane & 15;
    const int half   = lane >> 4;
    const int wid    = tid >> 5;

    v8f acc[2][8];
    #pragma unroll
    for (int i = 0; i < 2; ++i)
        #pragma unroll
        for (int j = 0; j < 8; ++j) acc[i][j] = {};

    #pragma unroll
    for (int kc = 0; kc < 128; kc += 32) {
        v16h a[2];
        #pragma unroll
        for (int rti = 0; rti < 2; ++rti) {
            const int rt  = wid * 2 + rti;            // row-tile 0..15
            const int blk = rt >> 2;                  // which weight matrix
            const float* W = (blk == 0) ? Wq : (blk == 1) ? Wk
                           : (blk == 2) ? Wv : Wv1;
            const int m = (rt * 16 + lane16) & 63;
            const float* wr = W + (size_t)m * C_;
            const float* p0 = wr + kc + half * 16;
            const float* p1 = wr + kc + half * 16 + 8;
            #pragma unroll
            for (int j = 0; j < 8; ++j) {
                a[rti][j]     = (_Float16)p0[j];
                a[rti][8 + j] = (_Float16)p1[j];
            }
        }
        #pragma unroll
        for (int ct = 0; ct < 8; ++ct) {
            v16h bf;
            const _Float16* bp = &Xs[(ct * 16 + lane16) * LDX + kc + half * 16];
            #pragma unroll
            for (int j = 0; j < 16; ++j) bf[j] = bp[j];
            acc[0][ct] = __builtin_amdgcn_wmma_f32_16x16x32_f16(
                false, a[0], false, bf, (short)0, acc[0][ct], false, false);
            acc[1][ct] = __builtin_amdgcn_wmma_f32_16x16x32_f16(
                false, a[1], false, bf, (short)0, acc[1][ct], false, false);
        }
    }

    #pragma unroll
    for (int rti = 0; rti < 2; ++rti) {
        const int rt  = wid * 2 + rti;
        const int blk = rt >> 2;
        #pragma unroll
        for (int ct = 0; ct < 8; ++ct) {
            const int Lg = l0 + ct * 16 + lane16;
            #pragma unroll
            for (int r = 0; r < 8; ++r) {
                const int cq  = (rt * 16 + r + 8 * half) & 63;
                const float v = acc[rti][ct][r];
                if (blk == 0)      Kh[((size_t)b * L_ + Lg) * CD + cq] = (_Float16)v;
                else if (blk == 1) Qh[((size_t)b * L_ + Lg) * CD + cq] = (_Float16)v;
                else if (blk == 2) Vh[((size_t)b * CD + cq) * L_ + Lg] = (_Float16)v;
                else               XV[((size_t)b * CD + cq) * L_ + Lg] = v;
            }
        }
    }
}

// -------------------------------------------------------------------------
// Kernel 2: fused flash attention, softmax over keys (= reference axis=1).
// out[b,c,m] = gamma * sum_l V[c,l] * softmax_l(K[:,l].Q[:,m]) + XV[b,c,m]
// K/V blocks staged once per WG into LDS via async-to-LDS, double buffered;
// each wave owns 16 queries; 8 WMMA per wave per 32-key block.
// Grid: (L/128, B), block 256 (8 waves).
// -------------------------------------------------------------------------

// LDS row pitches (f16 elements); all 16-B aligned rows.
#define LDK 72     // K block row: 64 d + pad     (144 B)
#define LDV 40     // V block row: 32 keys + pad  ( 80 B)
#define PLD 40     // P tile row : 32 keys + pad  ( 80 B)

// Stage one 32-key K/V block: K (32 x 64 f16) + V (64 x 32 f16), 256 threads,
// two 16-B chunks per thread -> 2 async instructions per wave per stage.
__device__ __forceinline__ void stage_kv(const _Float16* __restrict__ Kg,
                                         const _Float16* __restrict__ Vg,
                                         _Float16* __restrict__ Ksd,
                                         _Float16* __restrict__ Vsd,
                                         int kb, int tid) {
    {   // K: chunk tid -> key = tid>>3 (0..31), seg = tid&7 (8 f16 each)
        const int key = tid >> 3, seg = tid & 7;
        cp16_g2l(Kg + (size_t)(kb + key) * CD + seg * 8,
                 Ksd + key * LDK + seg * 8);
    }
    {   // V: chunk tid -> d = tid>>2 (0..63), seg = tid&3 (8 f16 each)
        const int d = tid >> 2, seg = tid & 3;
        cp16_g2l(Vg + (size_t)d * L_ + kb + seg * 8,
                 Vsd + d * LDV + seg * 8);
    }
}

__global__ __launch_bounds__(256) void attn_kernel(
    const _Float16* __restrict__ Qh, const _Float16* __restrict__ Kh,
    const _Float16* __restrict__ Vh, const float* __restrict__ XV,
    const float* __restrict__ gamma, float* __restrict__ out)
{
    __shared__ __align__(16) _Float16 Ks[2 * 32 * LDK];  //  9.0 KB
    __shared__ __align__(16) _Float16 Vs[2 * 64 * LDV];  // 10.0 KB
    __shared__ __align__(16) _Float16 Ps[8 * 16 * PLD];  // 10.0 KB

    const int b      = blockIdx.y;
    const int tid    = threadIdx.x;
    const int lane   = tid & 31;
    const int lane16 = lane & 15;
    const int half   = lane >> 4;
    const int wid    = tid >> 5;
    const int q0     = blockIdx.x * 128 + wid * 16;

    const _Float16* Kg = Kh + (size_t)b * L_ * CD;   // (L, 64)
    const _Float16* Vg = Vh + (size_t)b * CD * L_;   // (64, L)

    // Q A-fragments: 16 queries x 64 dims as two 16x32 tiles, kept in regs.
    v16h qa[2];
    {
        const _Float16* qp = Qh + ((size_t)b * L_ + q0 + lane16) * CD;
        #pragma unroll
        for (int c = 0; c < 2; ++c) {
            const _Float16* p0 = qp + c * 32 + half * 16;
            const _Float16* p1 = qp + c * 32 + half * 16 + 8;
            #pragma unroll
            for (int j = 0; j < 8; ++j) { qa[c][j] = p0[j]; qa[c][8 + j] = p1[j]; }
        }
    }

    v8f O[4];
    #pragma unroll
    for (int td = 0; td < 4; ++td) O[td] = {};
    float mrow[8], srow[8];
    #pragma unroll
    for (int r = 0; r < 8; ++r) { mrow[r] = -3.0e30f; srow[r] = 0.0f; }

    _Float16* ps = Ps + wid * 16 * PLD;

    // Prologue: stage block 0 into buffer 0.
    stage_kv(Kg, Vg, Ks, Vs, 0, tid);

    for (int kb = 0; kb < L_; kb += 32) {
        const int cur = (kb >> 5) & 1;
        _Float16* Ksc = Ks + cur * 32 * LDK;
        _Float16* Vsc = Vs + cur * 64 * LDV;

        // Prefetch next block into the other buffer; retire current block.
        if (kb + 32 < L_) {
            stage_kv(Kg, Vg, Ks + (cur ^ 1) * 32 * LDK,
                             Vs + (cur ^ 1) * 64 * LDV, kb + 32, tid);
            ASYNC_WAIT(2);        // keep only the 2 newest (next block) pending
        } else {
            ASYNC_WAIT(0);
        }
        __syncthreads();          // all waves' staged data visible

        // ---- S = Q . K^T  (2 tiles of 16 keys, K-dim 64 = 2 WMMA each) ----
        v8f S[2];
        #pragma unroll
        for (int t = 0; t < 2; ++t) {
            const _Float16* kp = Ksc + (t * 16 + lane16) * LDK;
            v16h b0, b1;
            #pragma unroll
            for (int j = 0; j < 16; ++j) {
                b0[j] = kp[half * 16 + j];           // d = 0..31 chunk
                b1[j] = kp[32 + half * 16 + j];      // d = 32..63 chunk
            }
            v8f z = {};
            v8f s0 = __builtin_amdgcn_wmma_f32_16x16x32_f16(
                false, qa[0], false, b0, (short)0, z, false, false);
            S[t] = __builtin_amdgcn_wmma_f32_16x16x32_f16(
                false, qa[1], false, b1, (short)0, s0, false, false);
        }

        // ---- Online softmax over the 32 new keys ----
        #pragma unroll
        for (int r = 0; r < 8; ++r) {
            float tm = fmaxf(S[0][r], S[1][r]);
            #pragma unroll
            for (int msk = 1; msk < 16; msk <<= 1)
                tm = fmaxf(tm, __shfl_xor(tm, msk, 32));  // within 16-lane half
            const float mnew  = fmaxf(mrow[r], tm);
            const float scale = __expf(mrow[r] - mnew);
            const float p0 = __expf(S[0][r] - mnew);
            const float p1 = __expf(S[1][r] - mnew);
            float rs = p0 + p1;
            #pragma unroll
            for (int msk = 1; msk < 16; msk <<= 1)
                rs += __shfl_xor(rs, msk, 32);
            srow[r] = srow[r] * scale + rs;
            mrow[r] = mnew;
            #pragma unroll
            for (int td = 0; td < 4; ++td) O[td][r] *= scale;
            // D-layout -> LDS (row = r + 8*half, col = key)
            ps[(r + 8 * half) * PLD + lane16]      = (_Float16)p0;
            ps[(r + 8 * half) * PLD + 16 + lane16] = (_Float16)p1;
        }
        __syncthreads();

        // ---- Re-read P in A-fragment layout (lane = query row) ----
        v16h pa;
        {
            const _Float16* pp = ps + lane16 * PLD;
            #pragma unroll
            for (int j = 0; j < 8; ++j) {
                pa[j]     = pp[half * 16 + j];
                pa[8 + j] = pp[half * 16 + 8 + j];
            }
        }
        __syncthreads();

        // ---- O += P . V  (4 d-tiles, one 16x16x32 WMMA each) ----
        #pragma unroll
        for (int td = 0; td < 4; ++td) {
            const _Float16* vp = Vsc + (td * 16 + lane16) * LDV + half * 16;
            v16h vb;
            #pragma unroll
            for (int j = 0; j < 16; ++j) vb[j] = vp[j];
            O[td] = __builtin_amdgcn_wmma_f32_16x16x32_f16(
                false, pa, false, vb, (short)0, O[td], false, false);
        }
        __syncthreads();          // protect buffers before next prefetch
    }

    // ---- Epilogue: out = gamma * O / l  + xv, transposed to (Cd, L) ----
    const float g = gamma[0];
    #pragma unroll
    for (int r = 0; r < 8; ++r) {
        const float inv = g / srow[r];            // srow >= 1
        const int   m   = q0 + r + 8 * half;
        #pragma unroll
        for (int td = 0; td < 4; ++td) {
            const int    c   = td * 16 + lane16;
            const size_t idx = ((size_t)b * CD + c) * L_ + m;
            out[idx] = O[td][r] * inv + XV[idx];
        }
    }
}

// -------------------------------------------------------------------------
extern "C" void kernel_launch(void* const* d_in, const int* in_sizes, int n_in,
                              void* d_out, int out_size, void* d_ws, size_t ws_size,
                              hipStream_t stream) {
    const float* x     = (const float*)d_in[0];
    const float* Wq    = (const float*)d_in[1];
    const float* Wk    = (const float*)d_in[2];
    const float* Wv    = (const float*)d_in[3];
    const float* Wv1   = (const float*)d_in[4];
    const float* gamma = (const float*)d_in[5];
    float* out = (float*)d_out;

    // Workspace: Kh | Qh | Vh (f16) then XV (f32). 24 MiB + 16 MiB = 40 MiB.
    const size_t nQ = (size_t)B_ * L_ * CD;
    _Float16* Kh = (_Float16*)d_ws;
    _Float16* Qh = Kh + nQ;
    _Float16* Vh = Qh + nQ;
    float*    XV = (float*)(Vh + nQ);

    dim3 blk(256);
    dim3 grid(L_ / 128, B_);
    proj_kernel<<<grid, blk, 0, stream>>>(x, Wq, Wk, Wv, Wv1, Kh, Qh, Vh, XV);
    attn_kernel<<<grid, blk, 0, stream>>>(Qh, Kh, Vh, XV, gamma, out);
}
